// DiscriminatorMADClf_81209241633290
// MI455X (gfx1250) — matrix operand order repaired and checked by reading, not compiled, not run on hardware
//
#include <hip/hip_runtime.h>
#include <hip/hip_bf16.h>
#include <math.h>

typedef __attribute__((ext_vector_type(16))) _Float16 v16h;
typedef __attribute__((ext_vector_type(8)))  _Float16 v8h;
typedef __attribute__((ext_vector_type(8)))  float    v8f;

#define TILE_M 128
#define TILE_N 64
#define TILE_K 32

// Issue one async 16-byte global->LDS copy (ASYNCcnt-tracked, no VGPR data).
__device__ __forceinline__ void async_copy_b128(const _Float16* g, const _Float16* lds)
{
    unsigned long long ga = (unsigned long long)(uintptr_t)g;
    unsigned           la = (unsigned)(uintptr_t)lds;   // low 32 bits = LDS offset
    asm volatile("global_load_async_to_lds_b128 %0, %1, off"
                 :: "v"(la), "v"(ga) : "memory");
}
__device__ __forceinline__ void async_wait0()
{
    asm volatile("s_wait_asynccnt 0x0" ::: "memory");
}

// C = A[M x K] * Wt^T (+ bias)(+ ELU). A: f16 row-major [M][K].
// Wt: f16 n-major [N][K] (pre-transposed weights). Outputs f16 and/or f32.
// Double-buffered async LDS staging: copy tile k+1 overlaps WMMA on tile k.
template<bool ELU, bool BIAS, bool O16, bool O32>
__global__ __launch_bounds__(256)
void gemm_wmma_f16(const _Float16* __restrict__ A, const _Float16* __restrict__ Wt,
                   const float* __restrict__ bias,
                   _Float16* __restrict__ C16, float* __restrict__ C32,
                   int Mdim, int Ndim, int Kdim)
{
    __shared__ alignas(64) _Float16 sA[2][TILE_M * TILE_K]; // 2 x 8 KB
    __shared__ alignas(64) _Float16 sB[2][TILE_N * TILE_K]; // 2 x 4 KB

    const int tid  = threadIdx.x;
    const int wave = tid >> 5;       // 0..7
    const int lane = tid & 31;
    const int hl   = lane >> 4;      // half-wave: 0 or 1
    const int l16  = lane & 15;

    const int blockRow = blockIdx.x * TILE_M;
    const int blockCol = blockIdx.y * TILE_N;
    const int waveRow  = wave * 16;  // 16-row strip per wave

    // Per-thread staging coordinates (fixed across iterations).
    const int aRow0 = tid >> 2;              // chunk 0: rows 0..63
    const int aRow1 = (tid + 256) >> 2;      // chunk 1: rows 64..127
    const int aKK   = (tid & 3) * 8;
    const int bN    = tid >> 2;
    const int bKK   = (tid & 3) * 8;

    auto stage = [&](int k0, int buf) {
        async_copy_b128(&A[(size_t)(blockRow + aRow0) * Kdim + k0 + aKK],
                        &sA[buf][aRow0 * TILE_K + aKK]);
        async_copy_b128(&A[(size_t)(blockRow + aRow1) * Kdim + k0 + aKK],
                        &sA[buf][aRow1 * TILE_K + aKK]);
        async_copy_b128(&Wt[(size_t)(blockCol + bN) * Kdim + k0 + bKK],
                        &sB[buf][bN * TILE_K + bKK]);
    };

    v8f acc[4] = { {}, {}, {}, {} };

    const int nk = Kdim / TILE_K;

    // Prologue: fill buffer 0.
    stage(0, 0);
    async_wait0();
    __syncthreads();

    for (int kt = 0; kt < nk; ++kt) {
        const int buf = kt & 1;

        // Kick off next tile into the other buffer (overlaps with WMMA below).
        if (kt + 1 < nk) stage((kt + 1) * TILE_K, buf ^ 1);

        // A fragment: row = waveRow + l16; elem e <-> K = hl*16 + e (contiguous).
        const _Float16* ap = &sA[buf][(waveRow + l16) * TILE_K + hl * 16];
        v8h alo = *(const v8h*)(ap);
        v8h ahi = *(const v8h*)(ap + 8);
        v16h afrag = __builtin_shufflevector(alo, ahi,
                        0,1,2,3,4,5,6,7,8,9,10,11,12,13,14,15);

        #pragma unroll
        for (int nt = 0; nt < 4; ++nt) {
            // B fragment: col n = nt*16 + l16, elem e <-> K = hl*16 + e (contiguous).
            const _Float16* bp = &sB[buf][(nt * 16 + l16) * TILE_K + hl * 16];
            v16h bfrag = *(const v16h*)(bp);
            acc[nt] = __builtin_amdgcn_wmma_f32_16x16x32_f16(
                false, afrag, false, bfrag, (short)0, acc[nt], false, false);
        }

        // Next tile must be fully in LDS (and this tile's reads done) before
        // the next iteration computes / restages.
        async_wait0();
        __syncthreads();
    }

    // Epilogue: C/D layout -> lane col N=l16, VGPR r row M = hl*8 + r.
    #pragma unroll
    for (int nt = 0; nt < 4; ++nt) {
        int col = blockCol + nt * 16 + l16;
        float bv = BIAS ? bias[col] : 0.0f;
        #pragma unroll
        for (int r = 0; r < 8; ++r) {
            int row = blockRow + waveRow + hl * 8 + r;
            float v = acc[nt][r] + bv;
            if (ELU) v = (v > 0.0f) ? v : expm1f(v);
            if (O16) C16[(size_t)row * Ndim + col] = (_Float16)v;
            if (O32) C32[(size_t)row * Ndim + col] = v;
        }
    }
}

// f32 -> f16 elementwise, 8 elems/thread.
__global__ __launch_bounds__(256)
void cvt_f16_kernel(const float* __restrict__ in, _Float16* __restrict__ out)
{
    int idx = blockIdx.x * 256 + threadIdx.x;
    const float4 f0 = ((const float4*)in)[idx * 2];
    const float4 f1 = ((const float4*)in)[idx * 2 + 1];
    v8h hv = { (_Float16)f0.x, (_Float16)f0.y, (_Float16)f0.z, (_Float16)f0.w,
               (_Float16)f1.x, (_Float16)f1.y, (_Float16)f1.z, (_Float16)f1.w };
    ((v8h*)out)[idx] = hv;
}

// W[K][Nc] f32 -> Wt[Nc][K] f16. Thread owns (n, 8-wide k chunk); reads are
// coalesced across lanes (consecutive tid -> consecutive n), writes b128.
__global__ __launch_bounds__(256)
void cvt_transpose_kernel(const float* __restrict__ W, _Float16* __restrict__ Wt,
                          int K, int Nc)
{
    int idx = blockIdx.x * 256 + threadIdx.x;  // K/8 * Nc chunks
    int k8  = idx / Nc;
    int n   = idx % Nc;
    v8h hv;
    #pragma unroll
    for (int j = 0; j < 8; ++j)
        hv[j] = (_Float16)W[(size_t)(k8 * 8 + j) * Nc + n];
    *(v8h*)(&Wt[(size_t)n * K + k8 * 8]) = hv;
}

// o_b[i,b] = sum_j exp(-|M[j,b]-M[i,b]|) - 1.  One block per i, thread = b.
#define MBD_TJ 64
__global__ __launch_bounds__(128)
void mbd_kernel(const float* __restrict__ M, float* __restrict__ ob, int N)
{
    __shared__ float sM[MBD_TJ * 128];
    const int b = threadIdx.x;
    const int i = blockIdx.x;
    const float mi = M[(size_t)i * 128 + b];
    float acc = 0.0f;
    for (int j0 = 0; j0 < N; j0 += MBD_TJ) {
        __syncthreads();
        #pragma unroll 4
        for (int r = 0; r < MBD_TJ; ++r)
            sM[r * 128 + b] = M[(size_t)(j0 + r) * 128 + b];
        __syncthreads();
        #pragma unroll 4
        for (int r = 0; r < MBD_TJ; ++r)
            acc += __expf(-fabsf(sM[r * 128 + b] - mi));
    }
    ob[(size_t)i * 128 + b] = acc - 1.0f;
}

// Tiny heads: mad_logits (17 cols) and clf_logits (10 cols).
__global__ __launch_bounds__(32)
void heads_kernel(const float* __restrict__ feat, const float* __restrict__ ob,
                  const float* __restrict__ Wm, const float* __restrict__ bm,
                  const float* __restrict__ Wc, const float* __restrict__ bc,
                  float* __restrict__ out_mad, float* __restrict__ out_clf)
{
    const int i = blockIdx.x;
    const int t = threadIdx.x;
    if (t < 17) {
        float acc = bm[t];
        for (int k = 0; k < 512; ++k)
            acc = fmaf(feat[(size_t)i * 512 + k], Wm[(size_t)k * 17 + t], acc);
        for (int b = 0; b < 128; ++b)
            acc = fmaf(ob[(size_t)i * 128 + b], Wm[(size_t)(512 + b) * 17 + t], acc);
        out_mad[(size_t)i * 17 + t] = acc;
    } else if (t < 27) {
        const int c = t - 17;
        float acc = bc[c];
        for (int k = 0; k < 512; ++k)
            acc = fmaf(feat[(size_t)i * 512 + k], Wc[(size_t)k * 10 + c], acc);
        out_clf[(size_t)i * 10 + c] = acc;
    }
}

extern "C" void kernel_launch(void* const* d_in, const int* in_sizes, int n_in,
                              void* d_out, int out_size, void* d_ws, size_t ws_size,
                              hipStream_t stream)
{
    const float* x   = (const float*)d_in[0];
    const float* Wb0 = (const float*)d_in[1];
    const float* bb0 = (const float*)d_in[2];
    const float* Wb1 = (const float*)d_in[3];
    const float* bb1 = (const float*)d_in[4];
    const float* T   = (const float*)d_in[5];
    const float* Wm  = (const float*)d_in[6];
    const float* bm  = (const float*)d_in[7];
    const float* Wc  = (const float*)d_in[8];
    const float* bc  = (const float*)d_in[9];

    const int N = 1024, DIN = 512, H = 1024, F = 512, B = 128;

    // Workspace layout (~9.3 MB)
    _Float16* x16  = (_Float16*)d_ws;                  // [N][DIN]
    _Float16* W0t  = x16 + (size_t)N * DIN;            // [H][DIN]
    _Float16* W1t  = W0t + (size_t)H * DIN;            // [F][H]
    _Float16* Tt   = W1t + (size_t)F * H;              // [B][F]
    _Float16* h16  = Tt  + (size_t)B * F;              // [N][H]
    _Float16* f16b = h16 + (size_t)N * H;              // [N][F]
    float* feat32  = (float*)(f16b + (size_t)N * F);   // [N][F]
    float* Mm      = feat32 + (size_t)N * F;           // [N][B]
    float* ob      = Mm + (size_t)N * B;               // [N][B]

    // ---- Prep: f16 conversion + weight transposes (one-time, bandwidth-trivial)
    cvt_f16_kernel<<<(N * DIN / 8) / 256, 256, 0, stream>>>(x, x16);
    cvt_transpose_kernel<<<(DIN / 8) * H / 256, 256, 0, stream>>>(Wb0, W0t, DIN, H);
    cvt_transpose_kernel<<<(H / 8) * F / 256, 256, 0, stream>>>(Wb1, W1t, H, F);
    cvt_transpose_kernel<<<(DIN / 8) * B / 256, 256, 0, stream>>>(T, Tt, DIN, B);

    // 1) h = elu(x @ Wb0 + bb0)  -> f16 only
    gemm_wmma_f16<true, true, true, false>
        <<<dim3(N / TILE_M, H / TILE_N), 256, 0, stream>>>(
            x16, W0t, bb0, h16, nullptr, N, H, DIN);
    // 2) feat = elu(h @ Wb1 + bb1) -> f16 (for GEMM3) + f32 (for heads)
    gemm_wmma_f16<true, true, true, true>
        <<<dim3(N / TILE_M, F / TILE_N), 256, 0, stream>>>(
            h16, W1t, bb1, f16b, feat32, N, F, H);
    // 3) M = feat @ T -> f32
    gemm_wmma_f16<false, false, false, true>
        <<<dim3(N / TILE_M, B / TILE_N), 256, 0, stream>>>(
            f16b, Tt, nullptr, nullptr, Mm, N, B, DIN);
    // 4) minibatch discrimination
    mbd_kernel<<<N, 128, 0, stream>>>(Mm, ob, N);
    // 5) heads -> d_out = [mad_logits (N x 17) | clf_logits (N x 10)]
    float* out_mad = (float*)d_out;
    float* out_clf = out_mad + (size_t)N * 17;
    heads_kernel<<<N, 32, 0, stream>>>(feat32, ob, Wm, bm, Wc, bc, out_mad, out_clf);
}
